// LSTM_27994596835467
// MI455X (gfx1250) — matrix operand order repaired and checked by reading, not compiled, not run on hardware
//
#include <hip/hip_runtime.h>
#include <hip/hip_bf16.h>

// Problem constants (from reference)
#define B_ 64
#define T_ 256
#define I_ 512
#define H_ 1024
#define G_ (4 * H_)   // 4096
#define C_ 10

typedef __bf16 bf16;
typedef __attribute__((ext_vector_type(16))) bf16  v16bf;
typedef __attribute__((ext_vector_type(8)))  bf16  v8bf;
typedef __attribute__((ext_vector_type(8)))  float v8f;

// ---------------- fp32 -> bf16 conversion ----------------
__global__ void cvt_f32_bf16(const float* __restrict__ src, bf16* __restrict__ dst, int n) {
    int i = blockIdx.x * blockDim.x + threadIdx.x;
    if (i < n) dst[i] = (bf16)src[i];
}

// ---------------- zero initial state ----------------
__global__ void init_state(bf16* __restrict__ h0, bf16* __restrict__ h1,
                           float* __restrict__ c0, float* __restrict__ c1) {
    int i = blockIdx.x * blockDim.x + threadIdx.x;
    if (i < B_ * H_) {
        h0[i] = (bf16)0.f;
        h1[i] = (bf16)0.f;
        c0[i] = 0.f;
        c1[i] = 0.f;
    }
}

// ---------------- WMMA fragment loaders (CDNA5 wave32 layouts) ----------------
// A: 16x32 bf16 (MxK). Lanes 0-15: row=lane, halfs = K {0..7, 16..23};
//                      lanes 16-31: row=lane-16, halfs = K {8..15, 24..31}.
__device__ __forceinline__ v16bf load_a_tile(const bf16* __restrict__ base, size_t ld,
                                             int m0, int k0, int lane) {
    const bf16* rowp = base + (size_t)(m0 + (lane & 15)) * ld;
    int kb = k0 + ((lane >> 4) << 3);            // +0 or +8
    union { v16bf v; v8bf h[2]; } u;
    u.h[0] = *(const v8bf*)(rowp + kb);          // K = kb .. kb+7
    u.h[1] = *(const v8bf*)(rowp + kb + 16);     // K = kb+16 .. kb+23
    return u.v;
}

// B: 32x16 bf16 (KxN) tile of W^T where W is [N_total, K] row-major:
// B[k][n] = W[n][k] -> per-lane two 8-bf16 runs along K (16-byte loads).
// Lanes 0-15: col=lane, K = {k0..k0+7, k0+16..k0+23}; lanes 16-31: col=lane-16, K = {k0+8..k0+15, k0+24..k0+31}.
__device__ __forceinline__ v16bf load_b_tile(const bf16* __restrict__ W, size_t ldw,
                                             int n0, int k0, int lane) {
    const bf16* p = W + (size_t)(n0 + (lane & 15)) * ldw + (size_t)(k0 + ((lane >> 4) << 3));
    union { v16bf v; v8bf h[2]; } u;
    u.h[0] = *(const v8bf*)p;
    u.h[1] = *(const v8bf*)(p + 16);
    return u.v;
}

__device__ __forceinline__ float sigmoidf_(float x) { return 1.f / (1.f + __expf(-x)); }

// ---------------- fused LSTM cell: gates GEMM (WMMA) + pointwise ----------------
// One wave owns a 16x16 (batch x hidden-unit) patch and computes all 4 gate
// tiles (cols j, H+j, 2H+j, 3H+j) so the A fragment is reused 4x and the cell
// nonlinearity runs on in-register accumulators.
// Launch: grid = 32 blocks, block = 256 threads (8 waves) -> 256 waves total
// = (B_/16) * (H_/16) tiles.
__global__ __launch_bounds__(256)
void lstm_cell_wmma(const bf16* __restrict__ inp, unsigned ld_inp, int Kx,  // [B_, Kx] (row stride ld_inp)
                    const bf16* __restrict__ Wx,                            // [G_, Kx]
                    const bf16* __restrict__ hprev,                         // [B_, H_]
                    const bf16* __restrict__ Wh,                            // [G_, H_]
                    const float* __restrict__ bx, const float* __restrict__ bh,
                    const float* __restrict__ cprev,                        // [B_, H_]
                    bf16* __restrict__ hout, float* __restrict__ cout) {
    int lane = threadIdx.x & 31;
    int wid  = blockIdx.x * (blockDim.x >> 5) + (threadIdx.x >> 5);
    int mt = wid >> 6;            // 0..3   (B_/16)
    int nt = wid & 63;            // 0..63  (H_/16)
    int m0 = mt << 4;
    int j0 = nt << 4;

    v8f acc[4] = {v8f{}, v8f{}, v8f{}, v8f{}};   // i, f, g, o

    // input contribution: inp @ Wx^T
    for (int k0 = 0; k0 < Kx; k0 += 32) {
        v16bf a = load_a_tile(inp, ld_inp, m0, k0, lane);
#pragma unroll
        for (int q = 0; q < 4; ++q) {
            v16bf b = load_b_tile(Wx, Kx, q * H_ + j0, k0, lane);
            acc[q] = __builtin_amdgcn_wmma_f32_16x16x32_bf16(
                false, a, false, b, (short)0, acc[q], false, false);
        }
    }
    // recurrent contribution: hprev @ Wh^T
    for (int k0 = 0; k0 < H_; k0 += 32) {
        v16bf a = load_a_tile(hprev, H_, m0, k0, lane);
#pragma unroll
        for (int q = 0; q < 4; ++q) {
            v16bf b = load_b_tile(Wh, H_, q * H_ + j0, k0, lane);
            acc[q] = __builtin_amdgcn_wmma_f32_16x16x32_bf16(
                false, a, false, b, (short)0, acc[q], false, false);
        }
    }

    // per-column biases (same column for all 8 accumulator rows in a lane)
    int j = j0 + (lane & 15);
    float bi  = bx[0 * H_ + j] + bh[0 * H_ + j];
    float bff = bx[1 * H_ + j] + bh[1 * H_ + j];
    float bg  = bx[2 * H_ + j] + bh[2 * H_ + j];
    float bo  = bx[3 * H_ + j] + bh[3 * H_ + j];

    // C/D layout: VGPR v, lanes 0-15 -> row m0+v; lanes 16-31 -> row m0+8+v
#pragma unroll
    for (int v = 0; v < 8; ++v) {
        int m = m0 + v + ((lane >> 4) << 3);
        float ig = sigmoidf_(acc[0][v] + bi);
        float fg = sigmoidf_(acc[1][v] + bff);
        float gg = tanhf(acc[2][v] + bg);
        float og = sigmoidf_(acc[3][v] + bo);
        float cp = cprev[(size_t)m * H_ + j];
        float cn = fg * cp + ig * gg;
        float hn = og * tanhf(cn);
        cout[(size_t)m * H_ + j] = cn;
        hout[(size_t)m * H_ + j] = (bf16)hn;
    }
}

// ---------------- final FC head: [64,1024] @ [1024->10] + bias ----------------
__global__ void fc_kernel(const bf16* __restrict__ h, const float* __restrict__ Wfc,
                          const float* __restrict__ bfc, float* __restrict__ out) {
    int idx = blockIdx.x * blockDim.x + threadIdx.x;   // 0..639
    if (idx >= B_ * C_) return;
    int b = idx / C_, c = idx % C_;
    float s = bfc[c];
    const bf16*  hp = h + (size_t)b * H_;
    const float* wp = Wfc + (size_t)c * H_;
    for (int k = 0; k < H_; ++k) s += (float)hp[k] * wp[k];
    out[idx] = s;   // out shape [1, B, C] flattened
}

extern "C" void kernel_launch(void* const* d_in, const int* in_sizes, int n_in,
                              void* d_out, int out_size, void* d_ws, size_t ws_size,
                              hipStream_t stream) {
    const float* x   = (const float*)d_in[0];
    const float* Wx0 = (const float*)d_in[1];
    const float* bx0 = (const float*)d_in[2];
    const float* Wh0 = (const float*)d_in[3];
    const float* bh0 = (const float*)d_in[4];
    const float* Wx1 = (const float*)d_in[5];
    const float* bx1 = (const float*)d_in[6];
    const float* Wh1 = (const float*)d_in[7];
    const float* bh1 = (const float*)d_in[8];
    const float* Wfc = (const float*)d_in[9];
    const float* bfc = (const float*)d_in[10];
    float* out = (float*)d_out;

    // workspace layout (all bf16 weights fit in the 192MB L2)
    char* ws = (char*)d_ws;
    size_t off = 0;
    auto take = [&](size_t bytes) -> char* {
        char* p = ws + off;
        off = (off + bytes + 255) & ~(size_t)255;
        return p;
    };
    bf16*  xb    = (bf16*)take((size_t)B_ * T_ * I_ * sizeof(bf16));  // 16.8 MB
    bf16*  Wx0b  = (bf16*)take((size_t)G_ * I_ * sizeof(bf16));       //  4.2 MB
    bf16*  Wh0b  = (bf16*)take((size_t)G_ * H_ * sizeof(bf16));       //  8.4 MB
    bf16*  Wx1b  = (bf16*)take((size_t)G_ * H_ * sizeof(bf16));
    bf16*  Wh1b  = (bf16*)take((size_t)G_ * H_ * sizeof(bf16));
    bf16*  h0buf[2] = { (bf16*)take((size_t)B_ * H_ * sizeof(bf16)),
                        (bf16*)take((size_t)B_ * H_ * sizeof(bf16)) };
    bf16*  h1buf[2] = { (bf16*)take((size_t)B_ * H_ * sizeof(bf16)),
                        (bf16*)take((size_t)B_ * H_ * sizeof(bf16)) };
    float* c0 = (float*)take((size_t)B_ * H_ * sizeof(float));
    float* c1 = (float*)take((size_t)B_ * H_ * sizeof(float));

    // 1) convert activations + weights to bf16 (re-done each call: deterministic)
    {
        int n = B_ * T_ * I_;
        cvt_f32_bf16<<<(n + 255) / 256, 256, 0, stream>>>(x, xb, n);
        n = G_ * I_;
        cvt_f32_bf16<<<(n + 255) / 256, 256, 0, stream>>>(Wx0, Wx0b, n);
        n = G_ * H_;
        cvt_f32_bf16<<<(n + 255) / 256, 256, 0, stream>>>(Wh0, Wh0b, n);
        cvt_f32_bf16<<<(n + 255) / 256, 256, 0, stream>>>(Wx1, Wx1b, n);
        cvt_f32_bf16<<<(n + 255) / 256, 256, 0, stream>>>(Wh1, Wh1b, n);
    }
    // 2) zero initial state (must happen every call; ws is not re-poisoned)
    {
        int n = B_ * H_;
        init_state<<<(n + 255) / 256, 256, 0, stream>>>(h0buf[0], h1buf[0], c0, c1);
    }

    // 3) sequential recurrence: 2 fused cell kernels per timestep.
    //    h is ping-pong buffered (all N-tiles read all of h_prev);
    //    c is updated in place (each wave owns its c elements).
    const int cellBlocks = (B_ / 16) * (H_ / 16) / 8;   // 32 blocks x 8 waves = 256 tiles
    int cur = 0;
    for (int t = 0; t < T_; ++t) {
        int nxt = cur ^ 1;
        // layer 0: input = x[:, t, :]  (row stride T_*I_)
        lstm_cell_wmma<<<cellBlocks, 256, 0, stream>>>(
            xb + (size_t)t * I_, (unsigned)(T_ * I_), I_, Wx0b,
            h0buf[cur], Wh0b, bx0, bh0, c0, h0buf[nxt], c0);
        // layer 1: input = new h0
        lstm_cell_wmma<<<cellBlocks, 256, 0, stream>>>(
            h0buf[nxt], (unsigned)H_, H_, Wx1b,
            h1buf[cur], Wh1b, bx1, bh1, c1, h1buf[nxt], c1);
        cur = nxt;
    }

    // 4) FC head on the final top-layer h
    fc_kernel<<<(B_ * C_ + 63) / 64, 64, 0, stream>>>(h1buf[cur], Wfc, bfc, out);
}
